// BinaryLayerWrapper_48550310314204
// MI455X (gfx1250) — 8-bit matrix operand order repaired and checked by reading, compiled, not run on hardware
//
#include <hip/hip_runtime.h>
#include <hip/hip_bf16.h>

typedef __attribute__((ext_vector_type(8))) int v8i;

#define B_    32
#define CIN   256
#define COUT  256
#define H_    56
#define W_    56
#define HW    (H_ * W_)
#define HP    58
#define WP    58
#define KTOT  (CIN * 9)
#define NCHUNK (KTOT / 64)
#define NTILE  (COUT / 16)
#define EPS   1e-5f

#define XB_BYTES  ((size_t)B_ * HP * WP * CIN)
#define PW_INTS   ((size_t)NTILE * NCHUNK * 32 * 8)
#define OFF_PWEI  (XB_BYTES)
#define OFF_SCALE (OFF_PWEI + PW_INTS * 4)
#define OFF_SHIFT (OFF_SCALE + 1024)
#define OFF_ALPHA (OFF_SHIFT + 1024)

__global__ void k_zero_xb(int4* __restrict__ xb) {
    size_t i = (size_t)blockIdx.x * blockDim.x + threadIdx.x;
    if (i < XB_BYTES / 16) xb[i] = make_int4(0, 0, 0, 0);
}

__global__ void k_bn_stats(const float* __restrict__ x,
                           const float* __restrict__ gamma,
                           const float* __restrict__ beta,
                           float* __restrict__ scale,
                           float* __restrict__ shift) {
    __shared__ float s1[256], s2[256];
    const int c = blockIdx.x;
    const int t = threadIdx.x;
    float sum = 0.f, sq = 0.f;
    for (int i = t; i < B_ * HW; i += 256) {
        int b = i / HW, hw = i % HW;
        float v = x[((size_t)b * CIN + c) * HW + hw];
        sum += v; sq += v * v;
    }
    s1[t] = sum; s2[t] = sq;
    __syncthreads();
    for (int s = 128; s > 0; s >>= 1) {
        if (t < s) { s1[t] += s1[t + s]; s2[t] += s2[t + s]; }
        __syncthreads();
    }
    if (t == 0) {
        float n = (float)(B_ * HW);
        float mean = s1[0] / n;
        float var  = s2[0] / n - mean * mean;
        float sc   = gamma[c] * rsqrtf(var + EPS);
        scale[c] = sc;
        shift[c] = beta[c] - mean * sc;
    }
}

__global__ void k_alpha(const float* __restrict__ w, float* __restrict__ alpha) {
    __shared__ float s1[256];
    const int co = blockIdx.x;
    const int t = threadIdx.x;
    float sum = 0.f;
    for (int i = t; i < CIN * 9; i += 256)
        sum += fabsf(w[(size_t)co * CIN * 9 + i]);
    s1[t] = sum;
    __syncthreads();
    for (int s = 128; s > 0; s >>= 1) {
        if (t < s) s1[t] += s1[t + s];
        __syncthreads();
    }
    if (t == 0) alpha[co] = s1[0] * (1.0f / (CIN * 9));
}

__global__ void k_pack_w(const float* __restrict__ w, int* __restrict__ pwei) {
    const int tile  = blockIdx.x / NCHUNK;
    const int chunk = blockIdx.x % NCHUNK;
    const int lane  = threadIdx.x >> 3;
    const int d     = threadIdx.x & 7;
    const int m  = lane & 15;
    const int lh = lane >> 4;
    const int co = tile * 16 + m;
    unsigned int packed = 0;
    #pragma unroll
    for (int byte = 0; byte < 4; ++byte) {
        int k  = 32 * lh + 4 * d + byte;
        int kg = chunk * 64 + k;
        int r  = kg >> 8;
        int ci = kg & 255;
        int kh = r / 3, kw = r % 3;
        float wv = w[(((size_t)co * CIN + ci) * 3 + kh) * 3 + kw];
        unsigned int s8 = (wv >= 0.f) ? 0x01u : 0xFFu;
        packed |= s8 << (8 * byte);
    }
    pwei[(((size_t)tile * NCHUNK + chunk) * 32 + lane) * 8 + d] = (int)packed;
}

__global__ void k_bn_pack(const float* __restrict__ x,
                          const float* __restrict__ scale,
                          const float* __restrict__ shift,
                          signed char* __restrict__ xb) {
    size_t i4 = (size_t)blockIdx.x * blockDim.x + threadIdx.x;
    size_t n4 = (size_t)B_ * CIN * HW / 4;
    if (i4 >= n4) return;
    int hw4 = (int)(i4 % (HW / 4));
    int bc  = (int)(i4 / (HW / 4));
    int c = bc % CIN, b = bc / CIN;
    int hw = hw4 * 4;
    int h = hw / W_, w0 = hw % W_;
    const float4 v = *(const float4*)(x + ((size_t)b * CIN + c) * HW + hw);
    float sc = scale[c], sh = shift[c];
    float f[4] = {v.x, v.y, v.z, v.w};
    size_t rowbase = (((size_t)b * HP + (h + 1)) * WP + (w0 + 1)) * CIN + c;
    #pragma unroll
    for (int j = 0; j < 4; ++j) {
        float xn = f[j] * sc + sh;
        xb[rowbase + (size_t)j * CIN] = (xn >= 0.f) ? (signed char)1 : (signed char)-1;
    }
}

__global__ void __launch_bounds__(256, 1)
k_gemm(const int* __restrict__ pwei,
       const signed char* __restrict__ xb,
       const float* __restrict__ alpha,
       float* __restrict__ out) {
    const int lane   = threadIdx.x & 31;
    const int waveId = threadIdx.x >> 5;
    const int ln = lane & 15;
    const int lh = lane >> 4;

    const int cp = waveId & 3;
    const int ph = waveId >> 2;
    const int tile0 = blockIdx.y * 8 + cp * 2;
    const int tile1 = tile0 + 1;
    const int b      = blockIdx.x / 49;
    const int hwBase = (blockIdx.x % 49) * 64 + ph * 32;

    const signed char *p0, *p1;
    {
        int hw, h, w;
        hw = hwBase + 0  + ln; h = hw / W_; w = hw % W_;
        p0 = xb + (((size_t)b * HP + h) * WP + w) * CIN + (size_t)lh * 32;
        hw = hwBase + 16 + ln; h = hw / W_; w = hw % W_;
        p1 = xb + (((size_t)b * HP + h) * WP + w) * CIN + (size_t)lh * 32;
    }
    const int* w0 = pwei + (((size_t)tile0 * NCHUNK) * 32 + lane) * 8;
    const int* w1 = pwei + (((size_t)tile1 * NCHUNK) * 32 + lane) * 8;

    v8i z = {0,0,0,0,0,0,0,0};
    v8i acc00 = z, acc01 = z;
    v8i acc10 = z, acc11 = z;

    #pragma unroll
    for (int chunk = 0; chunk < NCHUNK; ++chunk) {
        const int r   = chunk >> 2;
        const int cik = chunk & 3;
        const int dh  = r / 3;
        const int dw  = r % 3;
        const int boff = (dh * WP + dw) * CIN + cik * 64;
        const int woff = chunk * 256;

        v8i a0 = *(const v8i*)(w0 + woff);
        v8i a1 = *(const v8i*)(w1 + woff);
        v8i f0 = *(const v8i*)(p0 + boff);
        v8i f1 = *(const v8i*)(p1 + boff);

        acc00 = __builtin_amdgcn_wmma_i32_16x16x64_iu8(true, a0, true, f0, acc00, false, false);
        acc01 = __builtin_amdgcn_wmma_i32_16x16x64_iu8(true, a0, true, f1, acc01, false, false);
        acc10 = __builtin_amdgcn_wmma_i32_16x16x64_iu8(true, a1, true, f0, acc10, false, false);
        acc11 = __builtin_amdgcn_wmma_i32_16x16x64_iu8(true, a1, true, f1, acc11, false, false);
    }

    #pragma unroll
    for (int i = 0; i < 8; ++i) {
        int co0 = tile0 * 16 + lh * 8 + i;
        int co1 = tile1 * 16 + lh * 8 + i;
        float al0 = alpha[co0];
        float al1 = alpha[co1];
        size_t ob0 = ((size_t)b * COUT + co0) * HW + hwBase + ln;
        size_t ob1 = ((size_t)b * COUT + co1) * HW + hwBase + ln;
        out[ob0 +  0] = (float)acc00[i] * al0;
        out[ob0 + 16] = (float)acc01[i] * al0;
        out[ob1 +  0] = (float)acc10[i] * al1;
        out[ob1 + 16] = (float)acc11[i] * al1;
    }
}

extern "C" void kernel_launch(void* const* d_in, const int* in_sizes, int n_in,
                              void* d_out, int out_size, void* d_ws, size_t ws_size,
                              hipStream_t stream) {
    const float* x     = (const float*)d_in[0];
    const float* wgt   = (const float*)d_in[1];
    const float* gamma = (const float*)d_in[2];
    const float* beta  = (const float*)d_in[3];
    float* out = (float*)d_out;

    char* ws = (char*)d_ws;
    signed char* xb = (signed char*)ws;
    int*   pwei  = (int*)(ws + OFF_PWEI);
    float* scale = (float*)(ws + OFF_SCALE);
    float* shift = (float*)(ws + OFF_SHIFT);
    float* alpha = (float*)(ws + OFF_ALPHA);

    k_zero_xb<<<(unsigned)(XB_BYTES / 16 / 256), 256, 0, stream>>>((int4*)xb);
    k_bn_stats<<<CIN, 256, 0, stream>>>(x, gamma, beta, scale, shift);
    k_alpha<<<COUT, 256, 0, stream>>>(wgt, alpha);
    k_pack_w<<<NTILE * NCHUNK, 256, 0, stream>>>(wgt, pwei);
    {
        size_t n4 = (size_t)B_ * CIN * HW / 4;
        k_bn_pack<<<(unsigned)((n4 + 255) / 256), 256, 0, stream>>>(x, scale, shift, xb);
    }
    dim3 grid(32 * 49, 2);
    k_gemm<<<grid, 256, 0, stream>>>(pwei, xb, alpha, out);
}
